// AttentionHead_63977832841481
// MI455X (gfx1250) — matrix operand order repaired and checked by reading, not compiled, not run on hardware
//
#include <hip/hip_runtime.h>

typedef __attribute__((ext_vector_type(16))) __bf16 v16bf;
typedef __attribute__((ext_vector_type(8)))  float  v8f;
typedef __attribute__((ext_vector_type(8)))  __bf16 bf8v;
typedef __attribute__((ext_vector_type(4)))  float  f4v;

#define WMMA_BF16(A,B,Cc) __builtin_amdgcn_wmma_f32_16x16x32_bf16(false,(A),false,(B),(short)0,(Cc),false,false)

// CDNA5 async copy: global -> LDS, 16B per lane, tracked by ASYNCcnt.
__device__ __forceinline__ void async_ld16(unsigned lds_off, const void* g) {
  asm volatile("global_load_async_to_lds_b128 %0, %1, off"
               :: "v"(lds_off), "v"(g) : "memory");
}
#define WAIT_ASYNC(n) asm volatile("s_wait_asynccnt " #n ::: "memory")

// Load a 16x32 (or 32x16) bf16 WMMA operand: 16 contiguous halves per lane.
__device__ __forceinline__ v16bf ld_frag(const __bf16* p) {
  bf8v x0 = *(const bf8v*)p;
  bf8v x1 = *(const bf8v*)(p + 8);
  v16bf f;
  for (int i = 0; i < 8; i++) { f[i] = x0[i]; f[8 + i] = x1[i]; }
  return f;
}

static constexpr int Tn = 2048;   // tokens per batch
static constexpr int Cn = 1024;   // embed dim
static constexpr int DH = 64;     // head dim

// ---------------------------------------------------------------------------
// Kernel 0: convert Wq/Wk/Wv (fp32 [C,64]) -> bf16 transposed Wt[3][64][1024]
// ---------------------------------------------------------------------------
__global__ void wconv(const float* __restrict__ Wq, const float* __restrict__ Wk,
                      const float* __restrict__ Wv, __bf16* __restrict__ Wt) {
  int id  = blockIdx.x * blockDim.x + threadIdx.x;   // < 3*64*1024
  int mat = id >> 16;
  int rem = id & 65535;
  int n = rem >> 10, c = rem & 1023;
  const float* W = (mat == 0) ? Wq : ((mat == 1) ? Wk : Wv);
  Wt[id] = (__bf16)W[c * DH + n];
}

// ---------------------------------------------------------------------------
// Kernel 1: QKV projection.  Block = 4 waves x 16 tokens; weight K-slices are
// async-staged (double buffered) into LDS and shared by all 4 waves.
// ---------------------------------------------------------------------------
__global__ __launch_bounds__(128) void qkv_proj(
    const float* __restrict__ data, const __bf16* __restrict__ Wt,
    const float* __restrict__ bq, const float* __restrict__ bk,
    const float* __restrict__ bv,
    __bf16* __restrict__ Qb, __bf16* __restrict__ Kb, __bf16* __restrict__ Vt) {
  __shared__ __bf16 wsm[2 * 192 * 48];    // 2 bufs x (3*64 rows) x 48 halves (32 used)
  const int tid  = threadIdx.x;
  const int lane = tid & 31;
  const int wv   = tid >> 5;
  const int hi   = lane >> 4, lr = lane & 15;
  const int tile = blockIdx.x * 4 + wv;        // 16-token tile in flat [B*T]
  const long row0 = (long)tile * 16;
  const float* dptr = data + (row0 + lr) * Cn;
  const unsigned smbase = (unsigned)(size_t)(void*)wsm;

  // async-stage W[3][64][kk..kk+32) into LDS buffer bf (768 x 16B chunks)
  auto issueW = [&](int kk, int bf) {
    const char* Wb = (const char*)Wt + (size_t)kk * 2;
    unsigned dbase = smbase + (unsigned)bf * (192u * 96u);
    for (int i = 0; i < 6; i++) {
      int c = i * 128 + tid;
      int row = c >> 2, off = (c & 3) * 16;            // 4 chunks per 64B row slice
      async_ld16(dbase + (unsigned)row * 96u + off,
                 Wb + (size_t)row * 2048 + off);
    }
  };

  v8f acc[12];
  for (int i = 0; i < 12; i++) acc[i] = (v8f){0.f,0.f,0.f,0.f,0.f,0.f,0.f,0.f};

  issueW(0, 0);
  for (int it = 0; it < 32; it++) {
    const int kk = it * 32, bf = it & 1;
    if (it + 1 < 32) { issueW(kk + 32, bf ^ 1); WAIT_ASYNC(6); }
    else             { WAIT_ASYNC(0); }
    __syncthreads();

    // A fragment from global: lane row = lr, e -> K = hi*16 + e
    const float* ap = dptr + kk + hi * 16;
    f4v a0 = *(const f4v*)(ap);
    f4v a1 = *(const f4v*)(ap + 4);
    f4v a2 = *(const f4v*)(ap + 8);
    f4v a3 = *(const f4v*)(ap + 12);
    v16bf a;
    for (int i = 0; i < 4; i++) {
      a[i]      = (__bf16)a0[i];
      a[4 + i]  = (__bf16)a1[i];
      a[8 + i]  = (__bf16)a2[i];
      a[12 + i] = (__bf16)a3[i];
    }

    // Manual 2-deep software pipeline over the 12 B fragments: the fragment
    // consumed by WMMA m was loaded at step m-2, so each WMMA waits only on
    // older ds-loads while newer ones stay in flight.
    const __bf16* wl = wsm + bf * (192 * 48);
    auto wp = [&](int m) { return wl + (m * 16 + lr) * 48 + hi * 16; };
    v16bf f0 = ld_frag(wp(0));
    v16bf f1 = ld_frag(wp(1));
#pragma unroll
    for (int m = 0; m < 12; m++) {
      v16bf fn = (m + 2 < 12) ? ld_frag(wp(m + 2)) : f1;
      acc[m] = WMMA_BF16(a, f0, acc[m]);
      f0 = f1; f1 = fn;
    }
    __syncthreads();
  }

  const float scale = 0.03125f;               // 1024^-0.5, folded into Q
  const int b = tile >> 7;                    // 128 tiles per batch
  const int tloc0 = (tile & 127) * 16;
  for (int nt = 0; nt < 4; nt++) {
    int n = nt * 16 + lr;
    float bqv = bq[n], bkv = bk[n], bvv = bv[n];
    v8f cq = acc[nt], ck = acc[4 + nt], cv = acc[8 + nt];
    bf8v vpack;
    for (int r = 0; r < 8; r++) {
      long m = row0 + r + 8 * hi;             // C/D layout: row = r + 8*hi
      Qb[m * DH + n] = (__bf16)((cq[r] + bqv) * scale);
      Kb[m * DH + n] = (__bf16)(ck[r] + bkv);
      vpack[r] = (__bf16)(cv[r] + bvv);       // r -> t = tloc0 + 8*hi + r (contig)
    }
    *(bf8v*)(Vt + ((long)b * DH + n) * Tn + tloc0 + 8 * hi) = vpack;
  }
}

// ---------------------------------------------------------------------------
// Kernel 2: flash attention.  Block = 4 waves (same batch); K/V chunks are
// async-staged double-buffered into LDS and shared by the 4 waves.
// ---------------------------------------------------------------------------
__global__ __launch_bounds__(128) void attn(
    const __bf16* __restrict__ Qb, const __bf16* __restrict__ Kb,
    const __bf16* __restrict__ Vt, float* __restrict__ out) {
  // K: 2 x 128 rows x 72 halves (64 used)   = 36864 B
  // V: 2 x  64 rows x 136 halves (128 used) = 34816 B
  // P: 4 waves x 16 rows x 144 halves       = 18432 B
  __shared__ __bf16 smem[2 * 128 * 72 + 2 * 64 * 136 + 4 * 16 * 144];
  const int tid  = threadIdx.x;
  const int lane = tid & 31, wv = tid >> 5;
  const int hi = lane >> 4, lr = lane & 15;
  const int tile = blockIdx.x * 4 + wv;
  const long row0 = (long)tile * 16;
  const int b = tile >> 7;

  const __bf16* Kbase = Kb + (long)b * Tn * DH;
  const __bf16* Vbase = Vt + (long)b * DH * Tn;
  __bf16* P = smem + 2 * 128 * 72 + 2 * 64 * 136 + wv * (16 * 144);
  const unsigned smbase = (unsigned)(size_t)(void*)smem;
  const unsigned vldsbase = smbase + 36864u;

  // async-stage K[s0..s0+128) and V[:, s0..s0+128) into LDS buffer bf
  auto issueKV = [&](int s0, int bf) {
    const char* Ks = (const char*)Kbase + (size_t)s0 * 128;  // 128B per K row
    unsigned kd = smbase + (unsigned)bf * 18432u;
    for (int i = 0; i < 8; i++) {
      int c = i * 128 + tid;                    // 1024 x 16B chunks
      int row = c >> 3, off = (c & 7) * 16;
      async_ld16(kd + (unsigned)row * 144u + off, Ks + (size_t)row * 128 + off);
    }
    const char* Vs = (const char*)Vbase + (size_t)s0 * 2;
    unsigned vd = vldsbase + (unsigned)bf * 17408u;
    for (int i = 0; i < 8; i++) {
      int c = i * 128 + tid;                    // 1024 x 16B chunks
      int row = c >> 4, off = (c & 15) * 16;    // row = d, 256B per V row slice
      async_ld16(vd + (unsigned)row * 272u + off, Vs + (size_t)row * 4096 + off);
    }
  };

  // Q A-fragments for d in [0,32) and [32,64)
  v16bf aq[2];
  {
    const __bf16* qp = Qb + (row0 + lr) * DH;
    for (int f = 0; f < 2; f++) {
      bf8v q0 = *(const bf8v*)(qp + f * 32 + hi * 16);
      bf8v q1 = *(const bf8v*)(qp + f * 32 + 8 + hi * 16);
      for (int i = 0; i < 8; i++) { aq[f][i] = q0[i]; aq[f][8 + i] = q1[i]; }
    }
  }

  v8f o[4];
  for (int i = 0; i < 4; i++) o[i] = (v8f){0.f,0.f,0.f,0.f,0.f,0.f,0.f,0.f};
  float mrow[8], lrow[8];
  for (int r = 0; r < 8; r++) { mrow[r] = -3.0e38f; lrow[r] = 0.f; }

  issueKV(0, 0);
  for (int ch = 0; ch < Tn / 128; ch++) {
    const int s0 = ch * 128, bf = ch & 1;
    if (ch + 1 < Tn / 128) { issueKV(s0 + 128, bf ^ 1); WAIT_ASYNC(16); }
    else                   { WAIT_ASYNC(0); }
    __syncthreads();

    const __bf16* Kl = smem + bf * (128 * 72);
    const __bf16* Vl = smem + 2 * 128 * 72 + bf * (64 * 136);

    // ---- S = Q * K^T over this chunk: 1-tile-lookahead pipeline ----
    v8f s[8];
    {
      v16bf kb0 = ld_frag(Kl + lr * 72 + hi * 16);
      v16bf kb1 = ld_frag(Kl + lr * 72 + 32 + hi * 16);
#pragma unroll
      for (int j = 0; j < 8; j++) {
        v16bf nb0 = kb0, nb1 = kb1;
        if (j + 1 < 8) {
          const __bf16* kp = Kl + ((j + 1) * 16 + lr) * 72;
          nb0 = ld_frag(kp + hi * 16);        // d in [0,32)
          nb1 = ld_frag(kp + 32 + hi * 16);   // d in [32,64)
        }
        v8f z = (v8f){0.f,0.f,0.f,0.f,0.f,0.f,0.f,0.f};
        z = WMMA_BF16(aq[0], kb0, z);
        s[j] = WMMA_BF16(aq[1], kb1, z);
        kb0 = nb0; kb1 = nb1;
      }
    }

    // ---- online softmax (scale already folded into Q) ----
    float mnew[8], alpha[8], rs[8];
    for (int r = 0; r < 8; r++) {
      float cm = s[0][r];
      for (int j = 1; j < 8; j++) cm = fmaxf(cm, s[j][r]);
      for (int off = 1; off < 16; off <<= 1)
        cm = fmaxf(cm, __shfl_xor(cm, off, 32));   // reduce within 16-lane row-group
      mnew[r]  = fmaxf(mrow[r], cm);
      alpha[r] = __expf(mrow[r] - mnew[r]);
      rs[r]    = 0.f;
    }
    for (int j = 0; j < 8; j++)
      for (int r = 0; r < 8; r++) {
        float p = __expf(s[j][r] - mnew[r]);
        rs[r] += p;
        P[(r + 8 * hi) * 144 + j * 16 + lr] = (__bf16)p;
      }
    for (int r = 0; r < 8; r++) {
      for (int off = 1; off < 16; off <<= 1) rs[r] += __shfl_xor(rs[r], off, 32);
      lrow[r] = lrow[r] * alpha[r] + rs[r];
      mrow[r] = mnew[r];
    }
    for (int nt = 0; nt < 4; nt++)
      for (int r = 0; r < 8; r++) o[nt][r] *= alpha[r];

    asm volatile("s_wait_dscnt 0" ::: "memory");   // own-wave P writes visible

    // ---- O += P * V : flattened 16-step pipeline, 1-step B lookahead ----
    {
      auto vptr = [&](int t) {
        int j2 = t >> 2, nt = t & 3;
        return Vl + (nt * 16 + lr) * 136 + j2 * 32 + hi * 16;
      };
      auto pptr = [&](int j2) { return P + lr * 144 + j2 * 32 + hi * 16; };
      auto ldP = [&](int j2) {
        const __bf16* pp = pptr(j2);
        bf8v p0 = *(const bf8v*)pp;
        bf8v p1 = *(const bf8v*)(pp + 8);
        v16bf f;
        for (int i = 0; i < 8; i++) { f[i] = p0[i]; f[8 + i] = p1[i]; }
        return f;
      };
      v16bf apf = ldP(0), apn = apf;
      v16bf bcur = ld_frag(vptr(0));
#pragma unroll
      for (int t = 0; t < 16; t++) {
        const int nt = t & 3, j2 = t >> 2;
        v16bf bnext = bcur;
        if (t + 1 < 16) bnext = ld_frag(vptr(t + 1));
        if (nt == 0 && j2 + 1 < 4) apn = ldP(j2 + 1);   // P frag one k-slice ahead
        o[nt] = WMMA_BF16(apf, bcur, o[nt]);
        bcur = bnext;
        if (nt == 3) apf = apn;
      }
    }
    __syncthreads();    // protect buffer bf before next chunk's async writes
  }

  for (int nt = 0; nt < 4; nt++)
    for (int r = 0; r < 8; r++)
      out[(row0 + r + 8 * hi) * DH + nt * 16 + lr] = o[nt][r] / lrow[r];
}

// ---------------------------------------------------------------------------
extern "C" void kernel_launch(void* const* d_in, const int* in_sizes, int n_in,
                              void* d_out, int out_size, void* d_ws, size_t ws_size,
                              hipStream_t stream) {
  const float* data = (const float*)d_in[0];
  const float* Wq   = (const float*)d_in[1];
  const float* bq   = (const float*)d_in[2];
  const float* Wk   = (const float*)d_in[3];
  const float* bk   = (const float*)d_in[4];
  const float* Wv   = (const float*)d_in[5];
  const float* bv   = (const float*)d_in[6];

  char* ws = (char*)d_ws;
  const size_t WT_BYTES  = (size_t)3 * DH * Cn * 2;          // 384 KB
  const size_t QKV_BYTES = (size_t)8 * Tn * DH * 2;          // 2 MB each
  __bf16* Wt = (__bf16*)ws;
  __bf16* Qb = (__bf16*)(ws + WT_BYTES);
  __bf16* Kb = (__bf16*)(ws + WT_BYTES + QKV_BYTES);
  __bf16* Vt = (__bf16*)(ws + WT_BYTES + 2 * QKV_BYTES);

  hipLaunchKernelGGL(wconv, dim3(768), dim3(256), 0, stream, Wq, Wk, Wv, Wt);
  hipLaunchKernelGGL(qkv_proj, dim3(256), dim3(128), 0, stream,
                     data, Wt, bq, bk, bv, Qb, Kb, Vt);
  hipLaunchKernelGGL(attn, dim3(256), dim3(128), 0, stream,
                     Qb, Kb, Vt, (float*)d_out);
}
